// InteractionPPBlock_57707180589100
// MI455X (gfx1250) — matrix operand order repaired and checked by reading, not compiled, not run on hardware
//
#include <hip/hip_runtime.h>
#include <hip/hip_bf16.h>

typedef __attribute__((ext_vector_type(16))) _Float16 v16h;
typedef __attribute__((ext_vector_type(8)))  float    v8f;
typedef unsigned int v4u __attribute__((ext_vector_type(4)));
typedef int          v4i __attribute__((ext_vector_type(4)));
typedef int          v8i __attribute__((ext_vector_type(8)));

#define SR 136   // padded LDS row stride in halves (272B = 17*16B -> bank-rotating)

// f16 weight image offsets (in halves) inside the workspace weight buffer
#define WOFF_JI    0
#define WOFF_KJ    16384
#define WOFF_DOWN  32768      // [64][128]
#define WOFF_UP    40960      // [128][64]
#define WOFF_RB1   49152
#define WOFF_RB2   65536
#define WOFF_LIN   81920
#define WOFF_RA    98304      // ra1_0, ra2_0, ra1_1, ra2_1 (16384 each)
#define WBUF_HALVES 163840

#if defined(__has_builtin)
#if __has_builtin(__builtin_amdgcn_tensor_load_to_lds)
#define HAVE_TDM 1
#endif
#endif

__device__ __forceinline__ float silu(float v) {
    return v / (1.0f + __expf(-v));
}

// ---- Tensor Data Mover: 2-D f16 tile Global->LDS with row padding ----------
// D# per CDNA5 ISA.  rows = tensor_dim1/tile_dim1,
// hpr = halves per row (tensor_dim0/tile_dim0/stride).  pic/pac: pad codes.
#if HAVE_TDM
__device__ __forceinline__ void tdm_load_weight(const _Float16* gsrc, void* lds_dst,
                                                int rows, int hpr, int pic, int pac)
{
    unsigned long long ga = (unsigned long long)(__SIZE_TYPE__)gsrc;
    unsigned int lv = (unsigned int)(unsigned long long)(__SIZE_TYPE__)lds_dst; // LDS = addr[31:0]
    v4u g0;
    g0[0] = 1u;                                            // count=1, user descriptor
    g0[1] = lv;                                            // lds_addr
    g0[2] = (unsigned int)(ga & 0xFFFFFFFFu);              // global_addr[31:0]
    g0[3] = (unsigned int)((ga >> 32) & 0x1FFFFFFu)        // global_addr[56:32]
          | (2u << 30);                                    // type = 2 ("image")
    v8i g1;
    g1[0] = (1 << 16)                                      // data_size = 2 bytes
          | (1 << 20)                                      // pad_enable
          | (pic << 22)                                    // pad_interval
          | (pac << 25);                                   // pad_amount (DWORDs-1)
    g1[1] = (hpr & 0xFFFF) << 16;                          // tensor_dim0 lo16 @bit48
    g1[2] = ((hpr >> 16) & 0xFFFF) | ((rows & 0xFFFF) << 16); // dim0 hi / dim1 lo
    g1[3] = ((rows >> 16) & 0xFFFF) | ((hpr & 0xFFFF) << 16); // dim1 hi / tile_dim0
    g1[4] = (rows & 0xFFFF);                               // tile_dim1 (tile_dim2 = 0)
    g1[5] = hpr;                                           // tensor_dim0_stride[31:0]
    g1[6] = 0;                                             // stride hi / dim1_stride lo
    g1[7] = 0;
    v4i z4 = {0, 0, 0, 0};
#if __clang_major__ >= 23
    v8i z8 = {0, 0, 0, 0, 0, 0, 0, 0};
    __builtin_amdgcn_tensor_load_to_lds(g0, g1, z4, z4, z8, 0);
#else
    __builtin_amdgcn_tensor_load_to_lds(g0, g1, z4, z4, 0);
#endif
}
#define STAGE_W(dst, src, ROWS, HPR, PIC, PAC) \
    do { if (wave == 0) tdm_load_weight((src), (dst), (ROWS), (HPR), (PIC), (PAC)); } while (0)
#define WAIT_W() \
    do { if (wave == 0) __builtin_amdgcn_s_wait_tensorcnt(0); } while (0)
#else
#define STAGE_W(dst, src, ROWS, HPR, PIC, PAC)                                  \
    do { for (int _i = threadIdx.x; _i < (ROWS) * (HPR); _i += 256) {           \
             int _o = _i / (HPR), _k = _i % (HPR);                              \
             (dst)[_o * SR + _k] = (src)[_i]; } } while (0)
#define WAIT_W() do { } while (0)
#endif

// ---- WMMA fragment loaders -------------------------------------------------
// A-matrix 16x32 f16 layout (ISA 7.12.2): lane m = L&15 holds row m.
// lanes 0-15: K = 0..7 (v0-3), 16..23 (v4-7); lanes 16-31: K = 8..15, 24..31.
__device__ __forceinline__ v16h load_fragA(const _Float16* base, int row, int k0) {
    const int lane = threadIdx.x & 31;
    const _Float16* p = base + row * SR + k0 + ((lane >> 4) & 1) * 8;
    v16h a;
    ((uint4*)&a)[0] = *((const uint4*)p);        // K = kbase .. kbase+7
    ((uint4*)&a)[1] = *((const uint4*)(p + 16)); // K = kbase+16 .. kbase+23
    return a;
}

// B-matrix 32x16 f16 layout: lane n = L&15 holds column n.
// lanes 0-15: K = 0..7, 16..23; lanes 16-31: K = 8..15, 24..31 (matches A).
// sW holds W^T: sW[n*SR + k] = W[k][n].
__device__ __forceinline__ v16h load_fragB(const _Float16* baseT, int nrow, int k0) {
    const int lane = threadIdx.x & 31;
    const _Float16* p = baseT + nrow * SR + k0 + ((lane >> 4) & 1) * 8;
    v16h b;
    ((uint4*)&b)[0] = *((const uint4*)p);
    ((uint4*)&b)[1] = *((const uint4*)(p + 16));
    return b;
}

// ---- wave-level GEMM: 16 rows x (NT*16) cols, K = KT*32 --------------------
template<int KT, int NT>
__device__ __forceinline__ void wave_gemm(const _Float16* sA, const _Float16* sWT, v8f* acc) {
    const int lane = threadIdx.x & 31;
    const int wv = threadIdx.x >> 5;
    const int m = wv * 16 + (lane & 15);
#pragma unroll
    for (int kt = 0; kt < KT; ++kt) {
        v16h a = load_fragA(sA, m, kt * 32);
#pragma unroll
        for (int n = 0; n < NT; ++n) {
            v16h b = load_fragB(sWT, n * 16 + (lane & 15), kt * 32);
            acc[n] = __builtin_amdgcn_wmma_f32_16x16x32_f16(
                false, a, false, b, (short)0, acc[n], false, false);
        }
    }
}

template<int NT>
__device__ __forceinline__ void vclear(v8f* acc) {
#pragma unroll
    for (int n = 0; n < NT; ++n)
#pragma unroll
        for (int i = 0; i < 8; ++i) acc[n][i] = 0.0f;
}

// C/D layout (ISA 7.12.2): lane L, vgpr r -> row = 16*wave + (L>>4)*8 + r,
// col = (L&15) + 16*n
#define EPILOGUE_VARS()                                    \
    const int lane = threadIdx.x & 31;                     \
    const int wave = (int)(threadIdx.x >> 5);              \
    const int colb = lane & 15;                            \
    const int row0 = wave * 16 + ((lane >> 4) & 1) * 8;    \
    (void)lane; (void)wave; (void)colb; (void)row0;

// ===========================================================================
// Kernel 0: convert all weights once to transposed f16 images in workspace.
// dst[o*In + i] = (f16)W[i*Out + o]
// ===========================================================================
template<int In, int Out>
__device__ __forceinline__ void convT(const float* __restrict__ src,
                                      _Float16* __restrict__ dst, int tid, int nth) {
    for (int idx = tid; idx < In * Out; idx += nth) {
        int i = idx / Out, o = idx % Out;
        dst[o * In + i] = (_Float16)src[idx];
    }
}

__global__ __launch_bounds__(256) void k_prep_w(
    const float* __restrict__ W_ji, const float* __restrict__ W_kj,
    const float* __restrict__ W_down, const float* __restrict__ W_up,
    const float* __restrict__ rbW1, const float* __restrict__ rbW2,
    const float* __restrict__ W_lin,
    const float* __restrict__ raW1, const float* __restrict__ raW2,
    _Float16* __restrict__ wbuf)
{
    const int tid = blockIdx.x * 256 + threadIdx.x;
    const int nth = gridDim.x * 256;
    convT<128, 128>(W_ji,   wbuf + WOFF_JI,   tid, nth);
    convT<128, 128>(W_kj,   wbuf + WOFF_KJ,   tid, nth);
    convT<128, 64 >(W_down, wbuf + WOFF_DOWN, tid, nth);
    convT<64,  128>(W_up,   wbuf + WOFF_UP,   tid, nth);
    convT<128, 128>(rbW1,   wbuf + WOFF_RB1,  tid, nth);
    convT<128, 128>(rbW2,   wbuf + WOFF_RB2,  tid, nth);
    convT<128, 128>(W_lin,  wbuf + WOFF_LIN,  tid, nth);
    convT<128, 128>(raW1,           wbuf + WOFF_RA,          tid, nth);
    convT<128, 128>(raW2,           wbuf + WOFF_RA + 16384,  tid, nth);
    convT<128, 128>(raW1 + 16384,   wbuf + WOFF_RA + 32768,  tid, nth);
    convT<128, 128>(raW2 + 16384,   wbuf + WOFF_RA + 49152,  tid, nth);
}

// ===========================================================================
// Kernel 1: pre-aggregation.  x_ji = silu(x@W_ji+b); xk = silu(x@W_kj+b);
// rbf_t = (rbf@W_rbf1)@W_rbf2; xd = silu((xk*rbf_t)@W_down)
// ===========================================================================
__global__ __launch_bounds__(256) void k_pre(
    const float* __restrict__ x, const float* __restrict__ rbf,
    const float* __restrict__ W_rbf1, const float* __restrict__ W_rbf2,
    const float* __restrict__ b_kj, const float* __restrict__ b_ji,
    const _Float16* __restrict__ wbuf,
    float* __restrict__ xji_out, float* __restrict__ xd_out, int E)
{
    __shared__ _Float16 sX[128 * SR];
    __shared__ _Float16 sW[128 * SR];
    __shared__ _Float16 sT[128 * SR];
    __shared__ float    sB[128];
    __shared__ float    sR8[128 * 8];
    __shared__ float    sW2[8 * 128];

    const int e0 = blockIdx.x * 128;
    EPILOGUE_VARS();

    // kick off W_ji^T DMA, overlap with x-tile staging
    STAGE_W(sW, wbuf + WOFF_JI, 128, 128, 5, 3);
    for (int idx = threadIdx.x; idx < 128 * 128; idx += 256) {
        int r = idx >> 7, c = idx & 127;
        int ge = e0 + r;
        sX[r * SR + c] = (_Float16)((ge < E) ? x[ge * 128 + c] : 0.0f);
    }
    if (threadIdx.x < 128) sB[threadIdx.x] = b_ji[threadIdx.x];
    WAIT_W();
    __syncthreads();

    v8f acc[8];
    // ---- x_ji ----
    vclear<8>(acc);
    wave_gemm<4, 8>(sX, sW, acc);
#pragma unroll
    for (int n = 0; n < 8; ++n) {
        int col = colb + n * 16; float b = sB[col];
#pragma unroll
        for (int r = 0; r < 8; ++r) {
            int ge = e0 + row0 + r;
            if (ge < E) xji_out[ge * 128 + col] = silu(acc[n][r] + b);
        }
    }
    __syncthreads();

    // ---- x_kj * rbf_t ----
    STAGE_W(sW, wbuf + WOFF_KJ, 128, 128, 5, 3);
    if (threadIdx.x < 128) {
        sB[threadIdx.x] = b_kj[threadIdx.x];
        int r = threadIdx.x, ge = e0 + r;
        float rb[6];
#pragma unroll
        for (int i = 0; i < 6; ++i) rb[i] = (ge < E) ? rbf[ge * 6 + i] : 0.0f;
#pragma unroll
        for (int j = 0; j < 8; ++j) {
            float s = 0.0f;
#pragma unroll
            for (int i = 0; i < 6; ++i) s += rb[i] * W_rbf1[i * 8 + j];
            sR8[r * 8 + j] = s;
        }
    }
    for (int idx = threadIdx.x; idx < 8 * 128; idx += 256) sW2[idx] = W_rbf2[idx];
    WAIT_W();
    __syncthreads();

    vclear<8>(acc);
    wave_gemm<4, 8>(sX, sW, acc);
#pragma unroll
    for (int n = 0; n < 8; ++n) {
        int col = colb + n * 16; float b = sB[col];
#pragma unroll
        for (int r = 0; r < 8; ++r) {
            int row = row0 + r;
            float rt = 0.0f;
#pragma unroll
            for (int j = 0; j < 8; ++j) rt += sR8[row * 8 + j] * sW2[j * 128 + col];
            sT[row * SR + col] = (_Float16)(silu(acc[n][r] + b) * rt);
        }
    }
    __syncthreads();

    // ---- down projection to INT=64 ----
    STAGE_W(sW, wbuf + WOFF_DOWN, 64, 128, 5, 3);
    WAIT_W();
    __syncthreads();
    v8f acc4[4];
    vclear<4>(acc4);
    wave_gemm<4, 4>(sT, sW, acc4);
#pragma unroll
    for (int n = 0; n < 4; ++n) {
        int col = colb + n * 16;
#pragma unroll
        for (int r = 0; r < 8; ++r) {
            int ge = e0 + row0 + r;
            if (ge < E) xd_out[ge * 64 + col] = silu(acc4[n][r]);
        }
    }
}

// ===========================================================================
// Kernel 2: zero the aggregation buffer
// ===========================================================================
__global__ __launch_bounds__(256) void k_zero(float* __restrict__ p, long long n) {
    long long i = (long long)blockIdx.x * 256 + threadIdx.x;
    long long st = (long long)gridDim.x * 256;
    for (; i < n; i += st) p[i] = 0.0f;
}

// ===========================================================================
// Kernel 3: fused triplet message + scatter-add (wave per triplet).
// ===========================================================================
__global__ __launch_bounds__(256) void k_scatter(
    const float* __restrict__ sbf, const int* __restrict__ idx_kj,
    const int* __restrict__ idx_ji,
    const float* __restrict__ W_sbf1, const float* __restrict__ W_sbf2,
    const float* __restrict__ xd, float* __restrict__ aggp, int T)
{
    __shared__ float sW1[42 * 8];
    __shared__ float sW2[8 * 64];
    for (int i = threadIdx.x; i < 42 * 8; i += 256) sW1[i] = W_sbf1[i];
    for (int i = threadIdx.x; i < 8 * 64;  i += 256) sW2[i] = W_sbf2[i];
    __syncthreads();

    const int lane = threadIdx.x & 31;
    const int wv = threadIdx.x >> 5;
    const long long stride = (long long)gridDim.x * 8;

    for (long long t = (long long)blockIdx.x * 8 + wv; t < T; t += stride) {
        const int kj = idx_kj[t];
        const int ji = idx_ji[t];
        const float* srow = sbf + t * 42;
        __builtin_prefetch(sbf + (t + stride) * 42, 0, 1); // global_prefetch_b8

        float sj = 0.0f;
        if (lane < 8) {
#pragma unroll
            for (int i = 0; i < 42; ++i) sj += srow[i] * sW1[i * 8 + lane];
        }
        float s8[8];
#pragma unroll
        for (int j = 0; j < 8; ++j) s8[j] = __shfl(sj, j, 32);

#pragma unroll
        for (int cc = 0; cc < 2; ++cc) {
            int c = lane + cc * 32;
            float v = 0.0f;
#pragma unroll
            for (int j = 0; j < 8; ++j) v += s8[j] * sW2[j * 64 + c];
            v *= xd[(long long)kj * 64 + c];
            unsafeAtomicAdd(&aggp[(long long)ji * 64 + c], v);   // global_atomic_add_f32
        }
    }
}

// ===========================================================================
// Kernel 4: post-aggregation — 8 chained GEMM layers, h kept in LDS (f16),
// weights staged by TDM.
// ===========================================================================
__global__ __launch_bounds__(256) void k_post(
    const float* __restrict__ agg, const float* __restrict__ xji,
    const float* __restrict__ x,
    const float* __restrict__ rbb1, const float* __restrict__ rbb2,
    const float* __restrict__ b_lin,
    const float* __restrict__ rab1, const float* __restrict__ rab2,
    const _Float16* __restrict__ wbuf,
    float* __restrict__ out, int E)
{
    __shared__ _Float16 sA[128 * SR];
    __shared__ _Float16 sT[128 * SR];
    __shared__ _Float16 sW[128 * SR];
    __shared__ float    sB[128];

    const int e0 = blockIdx.x * 128;
    EPILOGUE_VARS();
    v8f acc[8];

    // L1: h = silu(agg @ W_up) + x_ji  -> sT     (W_up^T rows of 64 halves)
    STAGE_W(sW, wbuf + WOFF_UP, 128, 64, 4, 35);
    for (int idx = threadIdx.x; idx < 128 * 64; idx += 256) {
        int r = idx >> 6, c = idx & 63;
        int ge = e0 + r;
        sA[r * SR + c] = (_Float16)((ge < E) ? agg[ge * 64 + c] : 0.0f);
    }
    WAIT_W();
    __syncthreads();
    vclear<8>(acc);
    wave_gemm<2, 8>(sA, sW, acc);
#pragma unroll
    for (int n = 0; n < 8; ++n) {
        int col = colb + n * 16;
#pragma unroll
        for (int r = 0; r < 8; ++r) {
            int row = row0 + r, ge = e0 + row;
            float xj = (ge < E) ? xji[ge * 128 + col] : 0.0f;
            sT[row * SR + col] = (_Float16)(silu(acc[n][r]) + xj);
        }
    }
    __syncthreads();

    // L2: t1 = silu(h @ rbW1 + b1) -> sA
    STAGE_W(sW, wbuf + WOFF_RB1, 128, 128, 5, 3);
    if (threadIdx.x < 128) sB[threadIdx.x] = rbb1[threadIdx.x];
    WAIT_W();
    __syncthreads();
    vclear<8>(acc);
    wave_gemm<4, 8>(sT, sW, acc);
#pragma unroll
    for (int n = 0; n < 8; ++n) { int col = colb + n * 16; float b = sB[col];
#pragma unroll
        for (int r = 0; r < 8; ++r)
            sA[(row0 + r) * SR + col] = (_Float16)silu(acc[n][r] + b);
    }
    __syncthreads();

    // L3: h = h + silu(t1 @ rbW2 + b2) -> sT (per-element RMW by owning lane)
    STAGE_W(sW, wbuf + WOFF_RB2, 128, 128, 5, 3);
    if (threadIdx.x < 128) sB[threadIdx.x] = rbb2[threadIdx.x];
    WAIT_W();
    __syncthreads();
    vclear<8>(acc);
    wave_gemm<4, 8>(sA, sW, acc);
#pragma unroll
    for (int n = 0; n < 8; ++n) { int col = colb + n * 16; float b = sB[col];
#pragma unroll
        for (int r = 0; r < 8; ++r) { int row = row0 + r;
            float h = (float)sT[row * SR + col] + silu(acc[n][r] + b);
            sT[row * SR + col] = (_Float16)h;
        }
    }
    __syncthreads();

    // L4: h2 = silu(h @ W_lin + b_lin) + x -> sA
    STAGE_W(sW, wbuf + WOFF_LIN, 128, 128, 5, 3);
    if (threadIdx.x < 128) sB[threadIdx.x] = b_lin[threadIdx.x];
    WAIT_W();
    __syncthreads();
    vclear<8>(acc);
    wave_gemm<4, 8>(sT, sW, acc);
#pragma unroll
    for (int n = 0; n < 8; ++n) { int col = colb + n * 16; float b = sB[col];
#pragma unroll
        for (int r = 0; r < 8; ++r) { int row = row0 + r, ge = e0 + row;
            float xv = (ge < E) ? x[ge * 128 + col] : 0.0f;
            sA[row * SR + col] = (_Float16)(silu(acc[n][r] + b) + xv);
        }
    }
    __syncthreads();

    // Residual-after layers x2: (L5,L6) and (L7,L8)
#pragma unroll 1
    for (int layer = 0; layer < 2; ++layer) {
        const _Float16* w1 = wbuf + WOFF_RA + layer * 32768;
        const _Float16* w2 = w1 + 16384;
        const float* b1 = rab1 + layer * 128;
        const float* b2 = rab2 + layer * 128;

        // t = silu(h @ w1 + b1) -> sT
        STAGE_W(sW, w1, 128, 128, 5, 3);
        if (threadIdx.x < 128) sB[threadIdx.x] = b1[threadIdx.x];
        WAIT_W();
        __syncthreads();
        vclear<8>(acc);
        wave_gemm<4, 8>(sA, sW, acc);
#pragma unroll
        for (int n = 0; n < 8; ++n) { int col = colb + n * 16; float b = sB[col];
#pragma unroll
            for (int r = 0; r < 8; ++r)
                sT[(row0 + r) * SR + col] = (_Float16)silu(acc[n][r] + b);
        }
        __syncthreads();

        // h = h + silu(t @ w2 + b2)
        STAGE_W(sW, w2, 128, 128, 5, 3);
        if (threadIdx.x < 128) sB[threadIdx.x] = b2[threadIdx.x];
        WAIT_W();
        __syncthreads();
        vclear<8>(acc);
        wave_gemm<4, 8>(sT, sW, acc);
        if (layer == 0) {
#pragma unroll
            for (int n = 0; n < 8; ++n) { int col = colb + n * 16; float b = sB[col];
#pragma unroll
                for (int r = 0; r < 8; ++r) { int row = row0 + r;
                    float h = (float)sA[row * SR + col] + silu(acc[n][r] + b);
                    sA[row * SR + col] = (_Float16)h;
                }
            }
            __syncthreads();
        } else {
            // final: write f32 output
#pragma unroll
            for (int n = 0; n < 8; ++n) { int col = colb + n * 16; float b = sB[col];
#pragma unroll
                for (int r = 0; r < 8; ++r) { int row = row0 + r, ge = e0 + row;
                    if (ge < E)
                        out[ge * 128 + col] = (float)sA[row * SR + col] + silu(acc[n][r] + b);
                }
            }
        }
    }
}

// ===========================================================================
extern "C" void kernel_launch(void* const* d_in, const int* in_sizes, int n_in,
                              void* d_out, int out_size, void* d_ws, size_t ws_size,
                              hipStream_t stream)
{
    const float* x      = (const float*)d_in[0];
    const float* rbf    = (const float*)d_in[1];
    const float* sbf    = (const float*)d_in[2];
    const int*   idx_kj = (const int*)  d_in[3];
    const int*   idx_ji = (const int*)  d_in[4];
    const float* W_rbf1 = (const float*)d_in[5];
    const float* W_rbf2 = (const float*)d_in[6];
    const float* W_sbf1 = (const float*)d_in[7];
    const float* W_sbf2 = (const float*)d_in[8];
    const float* W_kj   = (const float*)d_in[9];
    const float* b_kj   = (const float*)d_in[10];
    const float* W_ji   = (const float*)d_in[11];
    const float* b_ji   = (const float*)d_in[12];
    const float* W_down = (const float*)d_in[13];
    const float* W_up   = (const float*)d_in[14];
    const float* rbW1   = (const float*)d_in[15];
    const float* rbb1   = (const float*)d_in[16];
    const float* rbW2   = (const float*)d_in[17];
    const float* rbb2   = (const float*)d_in[18];
    const float* W_lin  = (const float*)d_in[19];
    const float* b_lin  = (const float*)d_in[20];
    const float* raW1   = (const float*)d_in[21];
    const float* rab1   = (const float*)d_in[22];
    const float* raW2   = (const float*)d_in[23];
    const float* rab2   = (const float*)d_in[24];

    const int E = in_sizes[0] / 128;
    const int T = in_sizes[3];

    float* ws     = (float*)d_ws;
    float* ws_xji = ws;                          // E*128 f32
    float* ws_xd  = ws_xji + (size_t)E * 128;    // E*64  f32
    float* ws_agg = ws_xd  + (size_t)E * 64;     // E*64  f32
    _Float16* wbuf = (_Float16*)(ws_agg + (size_t)E * 64);  // WBUF_HALVES f16

    const int nblk = (E + 127) / 128;

    k_prep_w<<<64, 256, 0, stream>>>(W_ji, W_kj, W_down, W_up, rbW1, rbW2,
                                     W_lin, raW1, raW2, wbuf);
    k_pre<<<nblk, 256, 0, stream>>>(x, rbf, W_rbf1, W_rbf2, b_kj, b_ji,
                                    wbuf, ws_xji, ws_xd, E);
    k_zero<<<1024, 256, 0, stream>>>(ws_agg, (long long)E * 64);
    k_scatter<<<2048, 256, 0, stream>>>(sbf, idx_kj, idx_ji, W_sbf1, W_sbf2,
                                        ws_xd, ws_agg, T);
    k_post<<<nblk, 256, 0, stream>>>(ws_agg, ws_xji, x,
                                     rbb1, rbb2, b_lin, rab1, rab2,
                                     wbuf, (float*)d_out, E);
}
